// Transformer_Encoder_Layer_66494683677359
// MI455X (gfx1250) — matrix operand order repaired and checked by reading, not compiled, not run on hardware
//
#include <hip/hip_runtime.h>

typedef __attribute__((ext_vector_type(16))) __bf16 v16bf;
typedef __attribute__((ext_vector_type(8)))  float  v8f;
typedef unsigned int u32x4 __attribute__((ext_vector_type(4)));
typedef int          i32x4 __attribute__((ext_vector_type(4)));
typedef int          i32x8 __attribute__((ext_vector_type(8)));

#define CB 16
#define CS 384
#define CD 768
#define CH 12
#define CDH 64
#define CFF 3072

__device__ __forceinline__ v8f wmma_bf16(v16bf a, v16bf b, v8f c) {
  return __builtin_amdgcn_wmma_f32_16x16x32_bf16(false, a, false, b, (short)0, c, false, false);
}

#if __has_builtin(__builtin_amdgcn_tensor_load_to_lds)
#define HAVE_TDM 1
__device__ __forceinline__ void tdm_load(u32x4 g0, i32x8 g1) {
  i32x4 z4 = {0, 0, 0, 0};
#if __clang_major__ >= 23
  i32x8 z8 = {0, 0, 0, 0, 0, 0, 0, 0};
  __builtin_amdgcn_tensor_load_to_lds(g0, g1, z4, z4, z8, 0);
#else
  __builtin_amdgcn_tensor_load_to_lds(g0, g1, z4, z4, 0);
#endif
}
#else
#define HAVE_TDM 0
#endif

// Build TDM descriptor group0: count=1, lds byte addr, 57-bit global addr, type=2.
__device__ __forceinline__ u32x4 tdm_g0(unsigned lds_addr, const void* gptr) {
  unsigned long long ga = (unsigned long long)gptr;
  u32x4 g0 = {1u, lds_addr, (unsigned)ga,
              (unsigned)((ga >> 32) & 0x1FFFFFFu) | (2u << 30)};
  return g0;
}
// group1 for a 2D tile of bf16: dims/strides in elements.
__device__ __forceinline__ i32x8 tdm_g1(unsigned padcfg, unsigned td0, unsigned td1,
                                        unsigned tile0, unsigned tile1, unsigned stride0) {
  i32x8 g1 = {(int)(0x10000u | padcfg),                      // data_size=2B + padding cfg
              (int)((td0 & 0xFFFFu) << 16),                  // abar=0 | tensor_dim0 lo
              (int)(((td0 >> 16) & 0xFFFFu) | ((td1 & 0xFFFFu) << 16)),
              (int)(((td1 >> 16) & 0xFFFFu) | (tile0 << 16)),
              (int)(tile1 & 0xFFFFu),                        // tile_dim1 | tile_dim2=0
              (int)stride0,                                  // tensor_dim0_stride lo
              0, 0};
  return g1;
}

// ---------------- elementwise fp32 -> bf16 ----------------
__global__ void cvt_bf16_kernel(const float* __restrict__ in, __bf16* __restrict__ out, int n) {
  int i = blockIdx.x * blockDim.x + threadIdx.x;
  if (i < n) out[i] = (__bf16)in[i];
}

// qu = bf16(q + u[col]), qv = bf16(q + v[col])
__global__ void qbias_kernel(const float* __restrict__ q, const float* __restrict__ u,
                             const float* __restrict__ v, __bf16* __restrict__ qu,
                             __bf16* __restrict__ qv, int n) {
  int i = blockIdx.x * blockDim.x + threadIdx.x;
  if (i < n) {
    int c = i % CD;
    float x = q[i];
    qu[i] = (__bf16)(x + u[c]);
    qv[i] = (__bf16)(x + v[c]);
  }
}

// ---------------- tiled bf16 WMMA GEMM with TDM tile staging ----------------
// block = 128 threads (4 waves), tile 64x64, K-chunk 32. N,K mult of 64/32; M guarded
// (TDM OOB rows read as zero; epilogue store guarded).
__global__ __launch_bounds__(128)
void gemm_bf16_kernel(const __bf16* __restrict__ A, const __bf16* __restrict__ Bm,
                      float* outF, __bf16* outB, const float* bias,
                      int M, int N, int K, int relu)
{
  __shared__ __bf16 As[64][34];   // 32 elems/row + 1 DWORD TDM pad
  __shared__ __bf16 Bs[32][66];   // 64 elems/row + 1 DWORD TDM pad
  const int t    = threadIdx.x;
  const int lane = t & 31, w = t >> 5;
  const int wm   = w >> 1,  wn = w & 1;
  const int g    = lane >> 4, ln = lane & 15;
  const int row0 = blockIdx.y * 64;
  const int col0 = blockIdx.x * 64;

  v8f acc[2][2] = {};

  for (int k0 = 0; k0 < K; k0 += 32) {
#if HAVE_TDM
    if (w == 0) {
      // A tile 64x32: rows = 64B -> pad_interval code 3 (16 DW), pad 1 DW
      tdm_load(tdm_g0((unsigned)(size_t)(&As[0][0]), A + (size_t)row0 * K + k0),
               tdm_g1((1u << 20) | (3u << 22), 32u, (unsigned)(M - row0), 32u, 64u, (unsigned)K));
      // B tile 32x64: rows = 128B -> pad_interval code 4 (32 DW), pad 1 DW
      tdm_load(tdm_g0((unsigned)(size_t)(&Bs[0][0]), Bm + (size_t)k0 * N + col0),
               tdm_g1((1u << 20) | (4u << 22), 64u, 32u, 64u, 32u, (unsigned)N));
      __builtin_amdgcn_s_wait_tensorcnt(0);
    }
#else
    // cooperative fallback: 4-byte units
#pragma unroll
    for (int i = 0; i < 8; ++i) {
      int uu = i * 128 + t;
      { int r = uu >> 4, c = (uu & 15) * 2;
        unsigned tmp = 0;
        if (row0 + r < M) tmp = *(const unsigned*)(A + (size_t)(row0 + r) * K + k0 + c);
        *(unsigned*)(&As[r][c]) = tmp; }
      { int r = uu >> 5, c = (uu & 31) * 2;
        *(unsigned*)(&Bs[r][c]) = *(const unsigned*)(Bm + (size_t)(k0 + r) * N + col0 + c); }
    }
#endif
    __syncthreads();

    v16bf af[2], bfr[2];
#pragma unroll
    for (int ms = 0; ms < 2; ++ms) {
      int r = wm * 32 + ms * 16 + ln;
#pragma unroll
      for (int j = 0; j < 8; ++j) {
        int kk = 16 * g + 2 * j;
        af[ms][2 * j]     = As[r][kk];
        af[ms][2 * j + 1] = As[r][kk + 1];
      }
    }
#pragma unroll
    for (int ns = 0; ns < 2; ++ns) {
      int c = wn * 32 + ns * 16 + ln;
#pragma unroll
      for (int j = 0; j < 8; ++j) {
        int kk = 16 * g + 2 * j;
        bfr[ns][2 * j]     = Bs[kk][c];
        bfr[ns][2 * j + 1] = Bs[kk + 1][c];
      }
    }
#pragma unroll
    for (int ms = 0; ms < 2; ++ms)
#pragma unroll
      for (int ns = 0; ns < 2; ++ns)
        acc[ms][ns] = wmma_bf16(af[ms], bfr[ns], acc[ms][ns]);
    __syncthreads();
  }

#pragma unroll
  for (int ms = 0; ms < 2; ++ms)
#pragma unroll
    for (int ns = 0; ns < 2; ++ns) {
      int colb = col0 + wn * 32 + ns * 16 + ln;
      float bv = bias ? bias[colb] : 0.f;
#pragma unroll
      for (int v = 0; v < 8; ++v) {
        int r = row0 + wm * 32 + ms * 16 + v + 8 * g;
        if (r < M) {
          float x = acc[ms][ns][v] + bv;
          if (relu) x = fmaxf(x, 0.f);
          if (outF) outF[(size_t)r * N + colb] = x;
          if (outB) outB[(size_t)r * N + colb] = (__bf16)x;
        }
      }
    }
}

// ---------------- fused rel-pos flash attention ----------------
// grid = (S/64, B*H), block = 128 (4 waves); wave w owns q-rows [i0, i0+16).
__global__ __launch_bounds__(128)
void attn_kernel(const __bf16* __restrict__ qu, const __bf16* __restrict__ qv,
                 const __bf16* __restrict__ kb, const __bf16* __restrict__ vb,
                 const __bf16* __restrict__ rb, __bf16* __restrict__ attn_out)
{
  __shared__ float  Wl[4][16][49];
  __shared__ __bf16 pl[4][16][34];
  const int t    = threadIdx.x;
  const int lane = t & 31, w = t >> 5;
  const int g    = lane >> 4, ln = lane & 15;
  const int bh   = blockIdx.y;
  const int b    = bh / CH, h = bh % CH;
  const int i0   = blockIdx.x * 64 + w * 16;

  // load q+u / q+v A-fragments once (16 rows x 64 K)
  v16bf quf[2], qvf[2];
  size_t qbase = ((size_t)b * CS + i0) * CD + h * CDH;
#pragma unroll
  for (int hh = 0; hh < 2; ++hh)
#pragma unroll
    for (int j = 0; j < 8; ++j) {
      int kk = 32 * hh + 16 * g + 2 * j;
      size_t a = qbase + (size_t)ln * CD + kk;
      quf[hh][2 * j] = qu[a]; quf[hh][2 * j + 1] = qu[a + 1];
      qvf[hh][2 * j] = qv[a]; qvf[hh][2 * j + 1] = qv[a + 1];
    }

  v8f oacc[4] = {};
  float rm[8], rl[8];
#pragma unroll
  for (int v = 0; v < 8; ++v) { rm[v] = -1e30f; rl[v] = 0.f; }
  const float scale = 0.125f; // 1/sqrt(64)

  for (int j0 = 0; j0 < CS; j0 += 32) {
    // ---- content scores A_C: (q+u) @ k^T, two 16-col halves ----
    v8f sacc[2] = {};
#pragma unroll
    for (int hc = 0; hc < 2; ++hc) {
      int j = j0 + hc * 16 + ln;
      size_t kbase = ((size_t)b * CS + j) * CD + h * CDH;
#pragma unroll
      for (int hh = 0; hh < 2; ++hh) {
        v16bf kf;
#pragma unroll
        for (int jj = 0; jj < 8; ++jj) {
          int d = 32 * hh + 16 * g + 2 * jj;
          kf[2 * jj] = kb[kbase + d]; kf[2 * jj + 1] = kb[kbase + d + 1];
        }
        sacc[hc] = wmma_bf16(quf[hh], kf, sacc[hc]);
      }
    }

    // ---- relative band W[p,c] = (q+v)_{i0+p} . r[baseT+c], c in [0,48) ----
    int baseT = i0 - j0 - 31 + (CS - 1);
    v8f wacc[3] = {};
#pragma unroll
    for (int ns = 0; ns < 3; ++ns) {
      int c = ns * 16 + ln;
      int m = baseT + c;
      m = m < 0 ? 0 : (m > 2 * CS - 2 ? 2 * CS - 2 : m);
      size_t rbase = (size_t)m * CD + h * CDH;
#pragma unroll
      for (int hh = 0; hh < 2; ++hh) {
        v16bf rf;
#pragma unroll
        for (int jj = 0; jj < 8; ++jj) {
          int d = 32 * hh + 16 * g + 2 * jj;
          rf[2 * jj] = rb[rbase + d]; rf[2 * jj + 1] = rb[rbase + d + 1];
        }
        wacc[ns] = wmma_bf16(qvf[hh], rf, wacc[ns]);
      }
    }
#pragma unroll
    for (int ns = 0; ns < 3; ++ns)
#pragma unroll
      for (int v = 0; v < 8; ++v)
        Wl[w][v + 8 * g][ns * 16 + ln] = wacc[ns][v];
    __syncthreads();

    // ---- gather rel-shift, online softmax over the 32 new columns ----
    float pvv[2][8];
#pragma unroll
    for (int v = 0; v < 8; ++v) {
      int p = v + 8 * g;
      float sv[2], smax = -1e30f;
#pragma unroll
      for (int hc = 0; hc < 2; ++hc) {
        int qj = hc * 16 + ln;
        float s = (sacc[hc][v] + Wl[w][p][p - qj + 31]) * scale;
        sv[hc] = s;
        smax = fmaxf(smax, s);
      }
      for (int o = 1; o < 16; o <<= 1) smax = fmaxf(smax, __shfl_xor(smax, o, 32));
      float newm = fmaxf(rm[v], smax);
      float corr = __expf(rm[v] - newm);
      float rsum = 0.f;
#pragma unroll
      for (int hc = 0; hc < 2; ++hc) {
        float e = __expf(sv[hc] - newm);
        pvv[hc][v] = e; rsum += e;
      }
      for (int o = 1; o < 16; o <<= 1) rsum += __shfl_xor(rsum, o, 32);
      rl[v] = rl[v] * corr + rsum;
      rm[v] = newm;
#pragma unroll
      for (int ds = 0; ds < 4; ++ds) oacc[ds][v] *= corr;
    }
    __syncthreads();

    // ---- p (16x32) to LDS, reload as bf16 A-fragment ----
#pragma unroll
    for (int hc = 0; hc < 2; ++hc)
#pragma unroll
      for (int v = 0; v < 8; ++v)
        pl[w][v + 8 * g][hc * 16 + ln] = (__bf16)pvv[hc][v];
    __syncthreads();
    v16bf pf;
#pragma unroll
    for (int j = 0; j < 8; ++j) {
      int kk = 16 * g + 2 * j;
      pf[2 * j] = pl[w][ln][kk]; pf[2 * j + 1] = pl[w][ln][kk + 1];
    }

    // ---- o += p @ V (K=32 rows j0..j0+31) ----
#pragma unroll
    for (int ds = 0; ds < 4; ++ds) {
      v16bf vf;
#pragma unroll
      for (int jj = 0; jj < 8; ++jj) {
        int kk = 16 * g + 2 * jj;
        size_t vbase = ((size_t)b * CS + j0 + kk) * CD + h * CDH + ds * 16 + ln;
        vf[2 * jj] = vb[vbase]; vf[2 * jj + 1] = vb[vbase + CD];
      }
      oacc[ds] = wmma_bf16(pf, vf, oacc[ds]);
    }
    __syncthreads();
  }

  // normalize & store bf16 attention output [B,S,H*DH]
#pragma unroll
  for (int ds = 0; ds < 4; ++ds)
#pragma unroll
    for (int v = 0; v < 8; ++v) {
      int p = v + 8 * g;
      float o = oacc[ds][v] / rl[v];
      attn_out[((size_t)b * CS + i0 + p) * CD + h * CDH + ds * 16 + ln] = (__bf16)o;
    }
}

// ---------------- residual + LayerNorm ----------------
__global__ __launch_bounds__(256)
void add_ln_kernel(const float* __restrict__ X, const float* __restrict__ Y,
                   const float* __restrict__ gamma, const float* __restrict__ beta,
                   float* __restrict__ outF, __bf16* outB)
{
  __shared__ float red[256];
  const int row = blockIdx.x, t = threadIdx.x;
  const float* xr = X + (size_t)row * CD;
  const float* yr = Y + (size_t)row * CD;
  float s = 0.f;
  for (int c = t; c < CD; c += 256) s += xr[c] + yr[c];
  red[t] = s; __syncthreads();
  for (int o = 128; o > 0; o >>= 1) { if (t < o) red[t] += red[t + o]; __syncthreads(); }
  float mean = red[0] * (1.f / CD);
  __syncthreads();
  float s2 = 0.f;
  for (int c = t; c < CD; c += 256) { float d = xr[c] + yr[c] - mean; s2 += d * d; }
  red[t] = s2; __syncthreads();
  for (int o = 128; o > 0; o >>= 1) { if (t < o) red[t] += red[t + o]; __syncthreads(); }
  float inv = rsqrtf(red[0] * (1.f / CD) + 1e-5f);
  for (int c = t; c < CD; c += 256) {
    float vl = (xr[c] + yr[c] - mean) * inv * gamma[c] + beta[c];
    if (outF) outF[(size_t)row * CD + c] = vl;
    if (outB) outB[(size_t)row * CD + c] = (__bf16)vl;
  }
}

// ---------------- host orchestration ----------------
extern "C" void kernel_launch(void* const* d_in, const int* in_sizes, int n_in,
                              void* d_out, int out_size, void* d_ws, size_t ws_size,
                              hipStream_t stream) {
  (void)in_sizes; (void)n_in; (void)out_size; (void)ws_size;
  const float* x    = (const float*)d_in[0];
  const float* pe   = (const float*)d_in[1];
  const float* Wq   = (const float*)d_in[2];
  const float* Wk   = (const float*)d_in[3];
  const float* Wv   = (const float*)d_in[4];
  const float* Wr   = (const float*)d_in[5];
  const float* u    = (const float*)d_in[6];
  const float* vbia = (const float*)d_in[7];
  const float* Wo   = (const float*)d_in[8];
  const float* g1   = (const float*)d_in[9];
  const float* be1  = (const float*)d_in[10];
  const float* g2   = (const float*)d_in[11];
  const float* be2  = (const float*)d_in[12];
  const float* W1   = (const float*)d_in[13];
  const float* b1   = (const float*)d_in[14];
  const float* W2   = (const float*)d_in[15];
  const float* b2   = (const float*)d_in[16];
  float* out = (float*)d_out;

  const int n_x   = CB * CS * CD;
  const int n_pe  = (2 * CS - 1) * CD;
  const int n_W   = CD * CD;
  const int n_W1  = CD * CFF;
  const int n_hid = CB * CS * CFF;

  char* p = (char*)d_ws;
  auto alloc = [&](size_t bytes) -> void* {
    void* r = (void*)p; p += (bytes + 255) & ~(size_t)255; return r;
  };
  __bf16* x_bf    = (__bf16*)alloc((size_t)n_x * 2);
  __bf16* pe_bf   = (__bf16*)alloc((size_t)n_pe * 2);
  __bf16* Wq_bf   = (__bf16*)alloc((size_t)n_W * 2);
  __bf16* Wk_bf   = (__bf16*)alloc((size_t)n_W * 2);
  __bf16* Wv_bf   = (__bf16*)alloc((size_t)n_W * 2);
  __bf16* Wr_bf   = (__bf16*)alloc((size_t)n_W * 2);
  __bf16* Wo_bf   = (__bf16*)alloc((size_t)n_W * 2);
  __bf16* W1_bf   = (__bf16*)alloc((size_t)n_W1 * 2);
  __bf16* W2_bf   = (__bf16*)alloc((size_t)n_W1 * 2);
  float*  q_f     = (float*) alloc((size_t)n_x * 4);
  __bf16* qu_bf   = (__bf16*)alloc((size_t)n_x * 2);
  __bf16* qv_bf   = (__bf16*)alloc((size_t)n_x * 2);
  __bf16* k_bf    = (__bf16*)alloc((size_t)n_x * 2);
  __bf16* v_bf    = (__bf16*)alloc((size_t)n_x * 2);
  __bf16* r_bf    = (__bf16*)alloc((size_t)n_pe * 2);
  __bf16* attn_bf = (__bf16*)alloc((size_t)n_x * 2);
  float*  aproj_f = (float*) alloc((size_t)n_x * 4);
  float*  x1_f    = (float*) alloc((size_t)n_x * 4);
  __bf16* x1_bf   = (__bf16*)alloc((size_t)n_x * 2);
  __bf16* hid_bf  = (__bf16*)alloc((size_t)n_hid * 2);
  float*  ff_f    = (float*) alloc((size_t)n_x * 4);

  auto cvt = [&](const float* src, __bf16* dst, int n) {
    cvt_bf16_kernel<<<(n + 255) / 256, 256, 0, stream>>>(src, dst, n);
  };
  auto gemm = [&](const __bf16* A, const __bf16* Bm, float* oF, __bf16* oB,
                  const float* bias, int M, int N, int K, int relu) {
    dim3 gr(N / 64, (M + 63) / 64);
    gemm_bf16_kernel<<<gr, 128, 0, stream>>>(A, Bm, oF, oB, bias, M, N, K, relu);
  };

  // convert inputs/weights
  cvt(x, x_bf, n_x);
  cvt(pe, pe_bf, n_pe);
  cvt(Wq, Wq_bf, n_W);
  cvt(Wk, Wk_bf, n_W);
  cvt(Wv, Wv_bf, n_W);
  cvt(Wr, Wr_bf, n_W);
  cvt(Wo, Wo_bf, n_W);
  cvt(W1, W1_bf, n_W1);
  cvt(W2, W2_bf, n_W1);

  // projections
  gemm(x_bf, Wq_bf, q_f, nullptr, nullptr, CB * CS, CD, CD, 0);
  gemm(x_bf, Wk_bf, nullptr, k_bf, nullptr, CB * CS, CD, CD, 0);
  gemm(x_bf, Wv_bf, nullptr, v_bf, nullptr, CB * CS, CD, CD, 0);
  gemm(pe_bf, Wr_bf, nullptr, r_bf, nullptr, 2 * CS - 1, CD, CD, 0);
  qbias_kernel<<<(n_x + 255) / 256, 256, 0, stream>>>(q_f, u, vbia, qu_bf, qv_bf, n_x);

  // fused relative-position attention
  {
    dim3 gr(CS / 64, CB * CH);
    attn_kernel<<<gr, 128, 0, stream>>>(qu_bf, qv_bf, k_bf, v_bf, r_bf, attn_bf);
  }

  // output projection + residual/LN1
  gemm(attn_bf, Wo_bf, aproj_f, nullptr, nullptr, CB * CS, CD, CD, 0);
  add_ln_kernel<<<CB * CS, 256, 0, stream>>>(x, aproj_f, g1, be1, x1_f, x1_bf);

  // FFN
  gemm(x1_bf, W1_bf, nullptr, hid_bf, b1, CB * CS, CFF, CD, 1);
  gemm(hid_bf, W2_bf, ff_f, nullptr, b2, CB * CS, CD, CFF, 0);
  add_ln_kernel<<<CB * CS, 256, 0, stream>>>(x1_f, ff_f, g2, be2, out, nullptr);
}
